// NeuralSpectralBlock1d_4509715661385
// MI455X (gfx1250) — matrix operand order repaired and checked by reading, not compiled, not run on hardware
//
#include <hip/hip_runtime.h>
#include <math.h>

typedef __attribute__((ext_vector_type(16))) _Float16 v16h;
typedef __attribute__((ext_vector_type(8)))  float    v8f;

#define C_DIM 256
#define H_DIM 4096
#define HP_DIM 1024
#define NTC   64          // columns per workgroup
#define XH_STR 264        // padded K-stride (halves) of transposed f16 tile
#define PI_F  3.14159265358979f

// LDS layout (dynamic):
//   Xf : 256*64   f32 = 65536 B  (X tile, residual source)
//   Xh : 64*264   f16 = 33792 B  (transposed f16 X tile -> B fragments)
//   R  : 768*64   f16 = 98304 B  (GEMM result: rows 0..511 enc, 512..767 dec)
//   LT : 16*256*4 f32 = 65536 B  (latent tokens after spectral mixing)
#define SM_XF 0
#define SM_XH 65536
#define SM_R  (65536 + 33792)
#define SM_LT (65536 + 33792 + 98304)
#define SM_TOTAL (65536 + 33792 + 98304 + 65536)

__global__ __launch_bounds__(256, 1)
void nsb_fused(const float* __restrict__ x,
               const float* __restrict__ weights,   // (256, 24)
               const float* __restrict__ latent,    // (8, 4, 32)
               const float* __restrict__ enc_w,     // (512, 256)
               const float* __restrict__ enc_b,     // (512)
               const float* __restrict__ dec_w,     // (256, 256)
               const float* __restrict__ dec_b,     // (256)
               float* __restrict__ out)             // (32, 256, 4096)
{
  extern __shared__ unsigned char smem[];
  float*    Xf = (float*)(smem + SM_XF);
  _Float16* Xh = (_Float16*)(smem + SM_XH);
  _Float16* R  = (_Float16*)(smem + SM_R);
  float*    LT = (float*)(smem + SM_LT);

  const int tid  = threadIdx.x;
  const int lane = tid & 31;
  const int wave = tid >> 5;
  const int lhi  = lane >> 4;   // 0 or 1
  const int llo  = lane & 15;

  // ---------- phase 1: stage X tile (f32 + transposed f16) ----------
  // x is read exactly once overall -> non-temporal loads keep L2 free for W.
  {
    const int n     = tid & 63;
    const int c0    = (tid >> 6) * 64;             // contiguous 64-channel block
    const int nglob = blockIdx.x * NTC + n;
    const int b     = nglob >> 12;                 // / 4096
    const int hG    = nglob & (H_DIM - 1);
    const float* xb = x + (size_t)b * C_DIM * H_DIM + hG;
    for (int cc = 0; cc < 64; ++cc) {
      const int c = c0 + cc;
      const float v = __builtin_nontemporal_load(&xb[(size_t)c * H_DIM]);
      Xf[c * NTC + n]    = v;
      Xh[n * XH_STR + c] = (_Float16)v;
    }
  }
  __syncthreads();

  // ---------- phase 2: W(768x256) @ X(256x64) via WMMA f16 ----------
  // A fragments gathered straight from global (L2-resident weights);
  // B fragments from transposed f16 LDS tile. No barriers inside K-loop.
  v8f acc[6][4];
  const v8f vzero = {0.f, 0.f, 0.f, 0.f, 0.f, 0.f, 0.f, 0.f};
  #pragma unroll
  for (int mt = 0; mt < 6; ++mt)
    #pragma unroll
    for (int nt = 0; nt < 4; ++nt)
      acc[mt][nt] = vzero;

  // per-lane row pointers for this wave's 6 M-tiles
  const float* wrow[6];
  #pragma unroll
  for (int mt = 0; mt < 6; ++mt) {
    const int row = (wave * 6 + mt) * 16 + llo;
    wrow[mt] = (row < 512) ? (enc_w + (size_t)row * C_DIM)
                           : (dec_w + (size_t)(row - 512) * C_DIM);
  }

  #pragma unroll 2
  for (int kk = 0; kk < 8; ++kk) {
    const int K0 = kk * 32;

    // B fragments: 16 contiguous halves per lane -> 2x ds_load_b128 each
    v16h bfr[4];
    #pragma unroll
    for (int nt = 0; nt < 4; ++nt) {
      const _Float16* bp = Xh + (nt * 16 + llo) * XH_STR + K0 + lhi * 16;
      #pragma unroll
      for (int e = 0; e < 16; ++e) bfr[nt][e] = bp[e];
    }

    // A fragments: one contiguous 64B run per lane, direct from global
    #pragma unroll
    for (int mt = 0; mt < 6; ++mt) {
      const float* wp = wrow[mt] + K0 + 16 * lhi;
      if (kk < 7) __builtin_prefetch(wp + 32, 0, 3);   // next K-chunk row
      const float4 a0 = ((const float4*)wp)[0];
      const float4 a1 = ((const float4*)wp)[1];
      const float4 b0 = ((const float4*)(wp + 8))[0];
      const float4 b1 = ((const float4*)(wp + 8))[1];
      v16h afr;
      afr[0]  = (_Float16)a0.x; afr[1]  = (_Float16)a0.y;
      afr[2]  = (_Float16)a0.z; afr[3]  = (_Float16)a0.w;
      afr[4]  = (_Float16)a1.x; afr[5]  = (_Float16)a1.y;
      afr[6]  = (_Float16)a1.z; afr[7]  = (_Float16)a1.w;
      afr[8]  = (_Float16)b0.x; afr[9]  = (_Float16)b0.y;
      afr[10] = (_Float16)b0.z; afr[11] = (_Float16)b0.w;
      afr[12] = (_Float16)b1.x; afr[13] = (_Float16)b1.y;
      afr[14] = (_Float16)b1.z; afr[15] = (_Float16)b1.w;
      #pragma unroll
      for (int nt = 0; nt < 4; ++nt) {
        acc[mt][nt] = __builtin_amdgcn_wmma_f32_16x16x32_f16(
            false, afr, false, bfr[nt], (short)0, acc[mt][nt], false, false);
      }
    }
  }

  // ---------- store GEMM results (+bias) to LDS as f16 ----------
  #pragma unroll
  for (int mt = 0; mt < 6; ++mt) {
    #pragma unroll
    for (int r = 0; r < 8; ++r) {
      const int row = (wave * 6 + mt) * 16 + r + lhi * 8;  // C/D layout
      const float bias = (row < 512) ? enc_b[row] : dec_b[row - 512];
      #pragma unroll
      for (int nt = 0; nt < 4; ++nt)
        R[row * NTC + nt * 16 + llo] = (_Float16)(acc[mt][nt][r] + bias);
    }
  }
  __syncthreads();

  // ---------- phase 3a: latent cross-attn + spectral mixing ----------
  // thread = (patch 0..15, head 0..7, half-of-dh 0..1)
  const int pairId = tid >> 1;
  const int patch  = pairId >> 3;
  const int hh     = pairId & 7;
  const int hs     = tid & 1;
  const int rowK0  = hh * 64;        // enc rows: k = h*64+2d, v = h*64+2d+1
  const int nB     = patch * 4;

  float sc[4][4];                    // [token][patch-col]
  #pragma unroll
  for (int t = 0; t < 4; ++t)
    #pragma unroll
    for (int j = 0; j < 4; ++j) sc[t][j] = 0.f;
  for (int d = 0; d < 32; ++d) {
    float kj[4];
    #pragma unroll
    for (int j = 0; j < 4; ++j)
      kj[j] = (float)R[(rowK0 + 2 * d) * NTC + nB + j];
    #pragma unroll
    for (int t = 0; t < 4; ++t) {
      const float qv = latent[hh * 128 + t * 32 + d];
      #pragma unroll
      for (int j = 0; j < 4; ++j) sc[t][j] += qv * kj[j];
    }
  }
  #pragma unroll
  for (int t = 0; t < 4; ++t) {      // softmax over patch-cols
    const float mx = fmaxf(fmaxf(sc[t][0], sc[t][1]), fmaxf(sc[t][2], sc[t][3]));
    float s = 0.f;
    #pragma unroll
    for (int j = 0; j < 4; ++j) { sc[t][j] = __expf(sc[t][j] - mx); s += sc[t][j]; }
    const float inv = 1.f / s;
    #pragma unroll
    for (int j = 0; j < 4; ++j) sc[t][j] *= inv;
  }
  for (int dd = 0; dd < 16; ++dd) {
    const int d = hs * 16 + dd;
    const int c = hh * 32 + d;
    const float* wc = weights + c * 24;
    float vj[4];
    #pragma unroll
    for (int j = 0; j < 4; ++j)
      vj[j] = (float)R[(rowK0 + 2 * d + 1) * NTC + nB + j];
    #pragma unroll
    for (int t = 0; t < 4; ++t) {
      float val = latent[hh * 128 + t * 32 + d];       // +q residual
      #pragma unroll
      for (int j = 0; j < 4; ++j) val += sc[t][j] * vj[j];
      // spectral mixing: phase_m = m * theta; one sincos + angle-addition
      float sn1, cs1;
      __sincosf(val * (PI_F / 12.f), &sn1, &cs1);
      float sm = 0.f, cm = 1.f;                        // m = 0
      float a2 = val;                                  // +lt residual
      #pragma unroll
      for (int m = 0; m < 12; ++m) {
        a2 += sm * wc[m] + cm * wc[12 + m];
        const float sn_n = sm * cs1 + cm * sn1;
        const float cs_n = cm * cs1 - sm * sn1;
        sm = sn_n; cm = cs_n;
      }
      LT[(patch * 256 + c) * 4 + t] = a2;
    }
  }
  __syncthreads();

  // ---------- phase 3b: decoder cross-attn + residual + scatter-out ----------
  float s2[4][4];                    // [patch-col][token]
  #pragma unroll
  for (int j = 0; j < 4; ++j)
    #pragma unroll
    for (int t = 0; t < 4; ++t) s2[j][t] = 0.f;
  for (int d = 0; d < 32; ++d) {
    const int c = hh * 32 + d;
    float lv[4];
    #pragma unroll
    for (int t = 0; t < 4; ++t) lv[t] = LT[(patch * 256 + c) * 4 + t];
    #pragma unroll
    for (int j = 0; j < 4; ++j) {
      const float dq = (float)R[(512 + c) * NTC + nB + j];
      #pragma unroll
      for (int t = 0; t < 4; ++t) s2[j][t] += dq * lv[t];
    }
  }
  #pragma unroll
  for (int j = 0; j < 4; ++j) {      // softmax over tokens
    const float mx = fmaxf(fmaxf(s2[j][0], s2[j][1]), fmaxf(s2[j][2], s2[j][3]));
    float s = 0.f;
    #pragma unroll
    for (int t = 0; t < 4; ++t) { s2[j][t] = __expf(s2[j][t] - mx); s += s2[j][t]; }
    const float inv = 1.f / s;
    #pragma unroll
    for (int t = 0; t < 4; ++t) s2[j][t] *= inv;
  }

  const int nglobB = blockIdx.x * NTC + nB;
  const int b   = nglobB >> 12;
  const int hG  = nglobB & (H_DIM - 1);
  const int hp  = hG >> 2;
  // un-patchify exactly as torch: out[b, j*Hp*C + hp*C + c]
  float* outB = out + (size_t)b * C_DIM * H_DIM + (size_t)hp * C_DIM;
  for (int dd = 0; dd < 16; ++dd) {
    const int d = hs * 16 + dd;
    const int c = hh * 32 + d;
    float lv[4];
    #pragma unroll
    for (int t = 0; t < 4; ++t) lv[t] = LT[(patch * 256 + c) * 4 + t];
    #pragma unroll
    for (int j = 0; j < 4; ++j) {
      float o = 0.f;
      #pragma unroll
      for (int t = 0; t < 4; ++t) o += s2[j][t] * lv[t];
      o += Xf[c * NTC + nB + j];                       // +xp residual
      __builtin_nontemporal_store(o, &outB[(size_t)j * (HP_DIM * C_DIM) + c]);
    }
  }
}

extern "C" void kernel_launch(void* const* d_in, const int* in_sizes, int n_in,
                              void* d_out, int out_size, void* d_ws, size_t ws_size,
                              hipStream_t stream) {
  const float* x       = (const float*)d_in[0];
  const float* weights = (const float*)d_in[1];
  const float* latent  = (const float*)d_in[2];
  const float* enc_w   = (const float*)d_in[3];
  const float* enc_b   = (const float*)d_in[4];
  const float* dec_w   = (const float*)d_in[5];
  const float* dec_b   = (const float*)d_in[6];
  float* out = (float*)d_out;

  (void)in_sizes; (void)n_in; (void)out_size; (void)d_ws; (void)ws_size;

  const size_t smem = SM_TOTAL;  // 257 KB dynamic LDS (<= 320 KB/WGP)
  hipFuncSetAttribute((const void*)nsb_fused,
                      hipFuncAttributeMaxDynamicSharedMemorySize, (int)smem);
  const int ncols = 32 * 4096;         // B * H
  nsb_fused<<<ncols / NTC, 256, smem, stream>>>(
      x, weights, latent, enc_w, enc_b, dec_w, dec_b, out);
}
